// ShepherdEmbryo_77206332112984
// MI455X (gfx1250) — compile-verified, matrix operand order checked by reading, not run on hardware
//
#include <hip/hip_runtime.h>
#include <cstdint>
#include <cstddef>

#define D_MODEL 2048
#define NH      16
#define NKV     4
#define HD      128
#define KVD     (NKV * HD)     // 512
#define SEQ     2048
#define BATCH   2
#define FDIM    (4 * D_MODEL)  // 8192
#define MTOK    (BATCH * SEQ)  // 4096

typedef __attribute__((ext_vector_type(16))) __bf16 v16bf;
typedef __attribute__((ext_vector_type(8)))  float  v8f;
typedef __attribute__((ext_vector_type(4)))  unsigned int v4u;
typedef __attribute__((ext_vector_type(8)))  int v8i;
typedef __attribute__((ext_vector_type(4)))  int v4i;

#if defined(__has_builtin)
#if __has_builtin(__builtin_amdgcn_tensor_load_to_lds) && \
    __has_builtin(__builtin_amdgcn_s_wait_tensorcnt)
#define USE_TDM 1
#endif
#endif
#ifndef USE_TDM
#define USE_TDM 0
#endif

union Frag16 {
  v16bf v;
  uint4 q[2];
};

__device__ __forceinline__ unsigned short f32_to_bf16(float f) {
  unsigned int u = __float_as_uint(f);
  u += 0x7fffu + ((u >> 16) & 1u);   // round-to-nearest-even
  return (unsigned short)(u >> 16);
}

__device__ __forceinline__ v8f v8f_zero() {
  v8f z;
#pragma unroll
  for (int i = 0; i < 8; ++i) z[i] = 0.0f;
  return z;
}

#if USE_TDM
// ---------------------------------------------------------------------------
// Tensor Data Mover: DMA a 2D tile (tile_w x tile_h, 16-bit elems, row-major)
// from global into LDS with row padding. Descriptor per CDNA5 ISA 8.3-8.6.
// pad_interval_code: pad after 2^(c+1) DWORDs; pad_amount_code: (c+1) DWORDs.
// Issued by one wave; tracked by TENSORcnt.
// This toolchain's builtin takes 6 args (g0, g1, g2, g3, g4-unused, cpol).
// ---------------------------------------------------------------------------
__device__ __forceinline__ void tdm_load_2d(
    const void* gptr, unsigned int lds_off,
    unsigned int tile_w, unsigned int tile_h,
    unsigned int row_stride_elems,
    unsigned int tensor_w, unsigned int tensor_h,
    unsigned int pad_interval_code, unsigned int pad_amount_code) {
  unsigned long long ga = (unsigned long long)(size_t)gptr;
  v4u g0;
  g0[0] = 1u;                                            // count=1, user mode
  g0[1] = lds_off;                                       // lds_addr
  g0[2] = (unsigned int)(ga & 0xffffffffull);            // global_addr[31:0]
  g0[3] = (unsigned int)((ga >> 32) & 0x01ffffffull) |   // global_addr[56:32]
          (2u << 30);                                    // type=2 (image)
  v8i g1;
  g1[0] = (int)((1u << 16) |                             // data_size=2 bytes
                (1u << 20) |                             // pad_enable
                (pad_interval_code << 22) |
                (pad_amount_code << 25));
  g1[1] = (int)((tensor_w & 0xffffu) << 16);             // tensor_dim0[15:0]
  g1[2] = (int)((tensor_w >> 16) | ((tensor_h & 0xffffu) << 16));
  g1[3] = (int)((tensor_h >> 16) | (tile_w << 16));      // tile_dim0
  g1[4] = (int)tile_h;                                   // tile_dim1 (dim2=0)
  g1[5] = (int)row_stride_elems;                         // tensor_dim0_stride
  g1[6] = 0;
  g1[7] = 0;
  v4i z4;
  z4[0] = 0; z4[1] = 0; z4[2] = 0; z4[3] = 0;
  v8i z8;
#pragma unroll
  for (int i = 0; i < 8; ++i) z8[i] = 0;
  __builtin_amdgcn_tensor_load_to_lds(g0, g1, z4, z4, z8, 0);
}
#endif

// ---------------------------------------------------------------------------
// fp32 -> bf16 conversion (vectorized x4)
// ---------------------------------------------------------------------------
__global__ void k_cvt_bf16(const float* __restrict__ in,
                           unsigned short* __restrict__ out, int n4) {
  int i = blockIdx.x * blockDim.x + threadIdx.x;
  if (i < n4) {
    float4 f = ((const float4*)in)[i];
    ushort4 o;
    o.x = f32_to_bf16(f.x); o.y = f32_to_bf16(f.y);
    o.z = f32_to_bf16(f.z); o.w = f32_to_bf16(f.w);
    ((ushort4*)out)[i] = o;
  }
}

// ---------------------------------------------------------------------------
// Row RMSNorm: x[row,:] * rsqrt(mean(x^2)+eps) -> bf16
// ---------------------------------------------------------------------------
__global__ void k_rmsnorm_bf16(const float* __restrict__ x,
                               unsigned short* __restrict__ out,
                               int D, float eps) {
  int row = blockIdx.x;
  const float* xr = x + (size_t)row * D;
  float ss = 0.0f;
  for (int i = threadIdx.x; i < D; i += blockDim.x) {
    float v = xr[i];
    ss += v * v;
  }
#pragma unroll
  for (int m = 16; m; m >>= 1) ss += __shfl_xor(ss, m, 32);
  __shared__ float red[8];
  int wid = threadIdx.x >> 5, lid = threadIdx.x & 31;
  if (lid == 0) red[wid] = ss;
  __syncthreads();
  float tot = 0.0f;
#pragma unroll
  for (int i = 0; i < 8; ++i) tot += red[i];
  float sc = rsqrtf(tot / (float)D + eps);
  for (int i = threadIdx.x; i < D; i += blockDim.x)
    out[(size_t)row * D + i] = f32_to_bf16(xr[i] * sc);
}

// ---------------------------------------------------------------------------
// Shared compute stage for the GEMM: one 32-wide K slab from LDS tiles.
// ---------------------------------------------------------------------------
__device__ __forceinline__ void gemm_stage(const unsigned short* __restrict__ sA,
                                           const unsigned short* __restrict__ sB,
                                           v8f acc[4][2], int wm, int wn,
                                           int lm, int lg) {
  const int LDT = 40;
  Frag16 af[4], bf[2];
#pragma unroll
  for (int i = 0; i < 4; ++i) {
    const unsigned short* p = sA + (wm * 64 + i * 16 + lm) * LDT + lg * 8;
    af[i].q[0] = *(const uint4*)p;
    af[i].q[1] = *(const uint4*)(p + 16);
  }
#pragma unroll
  for (int j = 0; j < 2; ++j) {
    const unsigned short* p = sB + (wn * 32 + j * 16 + lm) * LDT + lg * 8;
    bf[j].q[0] = *(const uint4*)p;
    bf[j].q[1] = *(const uint4*)(p + 16);
  }
#pragma unroll
  for (int i = 0; i < 4; ++i)
#pragma unroll
    for (int j = 0; j < 2; ++j)
      acc[i][j] = __builtin_amdgcn_wmma_f32_16x16x32_bf16(
          false, af[i].v, false, bf[j].v, (short)0, acc[i][j], false, false);
}

// ---------------------------------------------------------------------------
// Generic bf16 WMMA GEMM: C[M,N] = A[M,K] @ B[N,K]^T   (row-major)
// 128x128 block tile, BK=32, 8 waves (2x4), each wave 64x32.
// TDM path: wave 0 DMA-loads tiles (with LDS padding) into double-buffered
// LDS while all waves compute the other buffer; one barrier per K slab.
// MODE 0: fp32   MODE 1: bf16(relu(c)^2)   MODE 2: fp32 resid + scale[col]*c
// ---------------------------------------------------------------------------
template <int MODE>
__global__ __launch_bounds__(256) void k_gemm_bf16(
    const unsigned short* __restrict__ A,
    const unsigned short* __restrict__ B,
    float* __restrict__ Cf, unsigned short* __restrict__ Cb,
    const float* __restrict__ resid, const float* __restrict__ scale,
    int M, int N, int K) {
  const int LDT = 40;  // padded row stride (elems) for a 32-elem K slab
  int bn = blockIdx.x * 128;
  int bm = blockIdx.y * 128;
  int t = threadIdx.x;
  int wave = t >> 5, lane = t & 31;
  int wm = wave & 1, wn = wave >> 1;
  int lg = lane >> 4, lm = lane & 15;

  v8f acc[4][2];
#pragma unroll
  for (int i = 0; i < 4; ++i)
#pragma unroll
    for (int j = 0; j < 2; ++j) acc[i][j] = v8f_zero();

  int nkb = K >> 5;

#if USE_TDM
  __shared__ __align__(16) unsigned short sA[2][128 * LDT];
  __shared__ __align__(16) unsigned short sB[2][128 * LDT];
  const unsigned int tileBytes = 128 * LDT * 2;
  unsigned int ldsA0 = (unsigned int)(size_t)&sA[0][0];
  unsigned int ldsB0 = (unsigned int)(size_t)&sB[0][0];
  if (wave == 0) {
    tdm_load_2d(A + (size_t)bm * K, ldsA0, 32, 128, K, K, M, 3, 3);
    tdm_load_2d(B + (size_t)bn * K, ldsB0, 32, 128, K, K, N, 3, 3);
  }
  for (int kb = 0; kb < nkb; ++kb) {
    int cur = kb & 1;
    if (wave == 0) __builtin_amdgcn_s_wait_tensorcnt(0);
    __syncthreads();  // stage kb visible; all waves done with stage kb-1
    if (wave == 0 && kb + 1 < nkb) {
      unsigned int nxt = (unsigned int)((kb + 1) & 1) * tileBytes;
      tdm_load_2d(A + (size_t)bm * K + (size_t)(kb + 1) * 32, ldsA0 + nxt,
                  32, 128, K, K, M, 3, 3);
      tdm_load_2d(B + (size_t)bn * K + (size_t)(kb + 1) * 32, ldsB0 + nxt,
                  32, 128, K, K, N, 3, 3);
    }
    gemm_stage(&sA[cur][0], &sB[cur][0], acc, wm, wn, lm, lg);
  }
#else
  __shared__ __align__(16) unsigned short sA[128 * LDT];
  __shared__ __align__(16) unsigned short sB[128 * LDT];
  for (int kb = 0; kb < nkb; ++kb) {
#pragma unroll
    for (int i = 0; i < 2; ++i) {
      int c = t + 256 * i;
      int row = c >> 2, c8 = c & 3;
      uint4 va = *(const uint4*)(A + (size_t)(bm + row) * K + kb * 32 + c8 * 8);
      *(uint4*)(sA + row * LDT + c8 * 8) = va;
      uint4 vb = *(const uint4*)(B + (size_t)(bn + row) * K + kb * 32 + c8 * 8);
      *(uint4*)(sB + row * LDT + c8 * 8) = vb;
    }
    __syncthreads();
    gemm_stage(sA, sB, acc, wm, wn, lm, lg);
    __syncthreads();
  }
#endif

#pragma unroll
  for (int i = 0; i < 4; ++i)
#pragma unroll
    for (int j = 0; j < 2; ++j) {
      int col = bn + wn * 32 + j * 16 + lm;
#pragma unroll
      for (int r = 0; r < 8; ++r) {
        int row = bm + wm * 64 + i * 16 + r + lg * 8;
        size_t idx = (size_t)row * N + col;
        float v = acc[i][j][r];
        if (MODE == 0) {
          Cf[idx] = v;
        } else if (MODE == 1) {
          float rl = v > 0.0f ? v : 0.0f;
          Cb[idx] = f32_to_bf16(rl * rl);
        } else {
          Cf[idx] = resid[idx] + scale[col] * v;
        }
      }
    }
}

// ---------------------------------------------------------------------------
// QK RMSNorm + RoPE + gain (fused), and V relayout; fp32 in -> bf16 out.
// Input  : [B*S, nheads*HD]   Output: [B, nheads, S, HD]
// mode 0: q (norm+rope+gain[h]/sqrt(HD)), 1: k (norm+rope), 2: v (copy)
// ---------------------------------------------------------------------------
__global__ void k_qkv_post(const float* __restrict__ in,
                           unsigned short* __restrict__ out,
                           const float* __restrict__ gain,
                           int nheads, int mode) {
  int vid = blockIdx.x * 8 + (threadIdx.x >> 5);
  int lane = threadIdx.x & 31;
  int row = vid / nheads;
  int h = vid - row * nheads;
  int b = row / SEQ, s = row - b * SEQ;
  const float* src = in + (size_t)row * (nheads * HD) + h * HD;
  float v0 = src[lane], v1 = src[lane + 32], v2 = src[lane + 64], v3 = src[lane + 96];
  if (mode < 2) {
    float ss = v0 * v0 + v1 * v1 + v2 * v2 + v3 * v3;
#pragma unroll
    for (int m = 16; m; m >>= 1) ss += __shfl_xor(ss, m, 32);
    float sc = rsqrtf(ss * (1.0f / HD) + 1.1920929e-07f);
    v0 *= sc; v1 *= sc; v2 *= sc; v3 *= sc;
    float f0 = powf(10000.0f, -((float)lane) / 64.0f);
    float f1 = powf(10000.0f, -((float)(lane + 32)) / 64.0f);
    float a0 = (float)s * f0, a1 = (float)s * f1;
    float c0 = __cosf(a0), s0 = __sinf(a0);
    float c1 = __cosf(a1), s1 = __sinf(a1);
    float o0 = v0 * c0 + v2 * s0;
    float o2 = -v0 * s0 + v2 * c0;
    float o1 = v1 * c1 + v3 * s1;
    float o3 = -v1 * s1 + v3 * c1;
    float g = (mode == 0) ? gain[h] * 0.08838834764831845f : 1.0f;  // 1/sqrt(128)
    v0 = o0 * g; v1 = o1 * g; v2 = o2 * g; v3 = o3 * g;
  }
  unsigned short* dst = out + ((size_t)(b * nheads + h) * SEQ + s) * HD;
  dst[lane] = f32_to_bf16(v0);
  dst[lane + 32] = f32_to_bf16(v1);
  dst[lane + 64] = f32_to_bf16(v2);
  dst[lane + 96] = f32_to_bf16(v3);
}

// ---------------------------------------------------------------------------
// Flash attention (causal, GQA). One block = 128 query rows of one (b,h).
// 8 waves x 16 q-rows each; K/V staged per 32-key block in LDS.
// K tile staged by TDM (wave 0); V transposed cooperatively.
// ---------------------------------------------------------------------------
__global__ __launch_bounds__(256) void k_attn(
    const unsigned short* __restrict__ Q,
    const unsigned short* __restrict__ Kk,
    const unsigned short* __restrict__ Vv,
    unsigned short* __restrict__ Y) {
  const int LKD = 136;  // K tile row stride (elems), conflict-free
  const int LVD = 40;   // V^T tile row stride
  const int LPD = 40;   // P scratch row stride
  __shared__ __align__(16) unsigned short sK[32 * LKD];
  __shared__ __align__(16) unsigned short sV[128 * LVD];     // [dim][key]
  __shared__ __align__(16) unsigned short sP[8 * 16 * LPD];  // per-wave

  int blk = blockIdx.x;
  int qt = blk & (SEQ / 128 - 1);
  int h = (blk / (SEQ / 128)) % NH;
  int b = blk / ((SEQ / 128) * NH);
  int kvh = h / (NH / NKV);
  int qbase = qt * 128;
  int t = threadIdx.x, wave = t >> 5, lane = t & 31;
  int lg = lane >> 4, lm = lane & 15;
  int qw = qbase + wave * 16;

  const unsigned short* qptr = Q + (size_t)(b * NH + h) * SEQ * HD;
  const unsigned short* kptr = Kk + (size_t)(b * NKV + kvh) * SEQ * HD;
  const unsigned short* vptr = Vv + (size_t)(b * NKV + kvh) * SEQ * HD;

  Frag16 qf[4];
#pragma unroll
  for (int kc = 0; kc < 4; ++kc) {
    const unsigned short* p = qptr + (size_t)(qw + lm) * HD + kc * 32 + lg * 8;
    qf[kc].q[0] = *(const uint4*)p;
    qf[kc].q[1] = *(const uint4*)(p + 16);
  }

  v8f yacc[8];
#pragma unroll
  for (int i = 0; i < 8; ++i) yacc[i] = v8f_zero();
  float rmax[8], rsum[8];
#pragma unroll
  for (int r = 0; r < 8; ++r) { rmax[r] = -3.0e38f; rsum[r] = 0.0f; }

  unsigned short* myP = sP + wave * 16 * LPD;
  int nkb = (qbase + 128) >> 5;

#if USE_TDM
  unsigned int ldsK0 = (unsigned int)(size_t)&sK[0];
#endif

  for (int kb = 0; kb < nkb; ++kb) {
    int kbase = kb * 32;
#if USE_TDM
    if (wave == 0)  // 32 keys x 128 dims, pad 64 DWORDs -> +4 DWORDs
      tdm_load_2d(kptr + (size_t)kbase * HD, ldsK0, 128, 32, HD, HD, SEQ, 5, 3);
#pragma unroll
    for (int i = 0; i < 2; ++i) {  // V transpose staging (all threads)
      int c = t + 256 * i;
      int key = c >> 4, d8 = c & 15;
      uint4 vv4 = *(const uint4*)(vptr + (size_t)(kbase + key) * HD + d8 * 8);
      const unsigned short* vs = (const unsigned short*)&vv4;
#pragma unroll
      for (int j = 0; j < 8; ++j) sV[(d8 * 8 + j) * LVD + key] = vs[j];
    }
    if (wave == 0) __builtin_amdgcn_s_wait_tensorcnt(0);
#else
#pragma unroll
    for (int i = 0; i < 2; ++i) {
      int c = t + 256 * i;
      int key = c >> 4, d8 = c & 15;
      uint4 kv4 = *(const uint4*)(kptr + (size_t)(kbase + key) * HD + d8 * 8);
      *(uint4*)(sK + key * LKD + d8 * 8) = kv4;
      uint4 vv4 = *(const uint4*)(vptr + (size_t)(kbase + key) * HD + d8 * 8);
      const unsigned short* vs = (const unsigned short*)&vv4;
#pragma unroll
      for (int j = 0; j < 8; ++j) sV[(d8 * 8 + j) * LVD + key] = vs[j];
    }
#endif
    __syncthreads();

    if (kbase <= qw + 15) {  // wave-uniform: EXEC stays all-ones for WMMA
      v8f st[2];
      st[0] = v8f_zero();
      st[1] = v8f_zero();
#pragma unroll
      for (int tt = 0; tt < 2; ++tt)
#pragma unroll
        for (int kc = 0; kc < 4; ++kc) {
          Frag16 kf;
          const unsigned short* p = sK + (tt * 16 + lm) * LKD + kc * 32 + lg * 8;
          kf.q[0] = *(const uint4*)p;
          kf.q[1] = *(const uint4*)(p + 16);
          st[tt] = __builtin_amdgcn_wmma_f32_16x16x32_bf16(
              false, qf[kc].v, false, kf.v, (short)0, st[tt], false, false);
        }
      float lmax[8];
#pragma unroll
      for (int r = 0; r < 8; ++r) lmax[r] = -3.0e38f;
#pragma unroll
      for (int tt = 0; tt < 2; ++tt)
#pragma unroll
        for (int r = 0; r < 8; ++r) {
          int qi = qw + r + 8 * lg;
          int kj = kbase + tt * 16 + lm;
          float v = st[tt][r];
          if (kj > qi) v = -3.0e38f;
          st[tt][r] = v;
          lmax[r] = fmaxf(lmax[r], v);
        }
#pragma unroll
      for (int r = 0; r < 8; ++r)
#pragma unroll
        for (int m = 1; m < 16; m <<= 1)
          lmax[r] = fmaxf(lmax[r], __shfl_xor(lmax[r], m, 32));
      float alpha[8];
#pragma unroll
      for (int r = 0; r < 8; ++r) {
        float nm = fmaxf(rmax[r], lmax[r]);
        alpha[r] = __expf(rmax[r] - nm);
        rmax[r] = nm;
      }
      float lsum[8];
#pragma unroll
      for (int r = 0; r < 8; ++r) lsum[r] = 0.0f;
#pragma unroll
      for (int tt = 0; tt < 2; ++tt)
#pragma unroll
        for (int r = 0; r < 8; ++r) {
          float p = __expf(st[tt][r] - rmax[r]);
          st[tt][r] = p;
          lsum[r] += p;
        }
#pragma unroll
      for (int r = 0; r < 8; ++r)
#pragma unroll
        for (int m = 1; m < 16; m <<= 1) lsum[r] += __shfl_xor(lsum[r], m, 32);
#pragma unroll
      for (int r = 0; r < 8; ++r) rsum[r] = rsum[r] * alpha[r] + lsum[r];
#pragma unroll
      for (int i = 0; i < 8; ++i)
#pragma unroll
        for (int r = 0; r < 8; ++r) yacc[i][r] *= alpha[r];
#pragma unroll
      for (int tt = 0; tt < 2; ++tt)
#pragma unroll
        for (int r = 0; r < 8; ++r)
          myP[(r + 8 * lg) * LPD + tt * 16 + lm] = f32_to_bf16(st[tt][r]);
      Frag16 pf;
      {
        const unsigned short* p = myP + lm * LPD + lg * 8;
        pf.q[0] = *(const uint4*)p;
        pf.q[1] = *(const uint4*)(p + 16);
      }
#pragma unroll
      for (int ni = 0; ni < 8; ++ni) {
        Frag16 vf;
        const unsigned short* p = sV + (ni * 16 + lm) * LVD + lg * 8;
        vf.q[0] = *(const uint4*)p;
        vf.q[1] = *(const uint4*)(p + 16);
        yacc[ni] = __builtin_amdgcn_wmma_f32_16x16x32_bf16(
            false, pf.v, false, vf.v, (short)0, yacc[ni], false, false);
      }
    }
    __syncthreads();
  }

#pragma unroll
  for (int ni = 0; ni < 8; ++ni) {
    int dcol = ni * 16 + lm;
#pragma unroll
    for (int r = 0; r < 8; ++r) {
      int qi = qw + r + 8 * lg;
      float val = yacc[ni][r] / rsum[r];
      size_t idx = (((size_t)b * SEQ + qi) * NH + h) * HD + dcol;
      Y[idx] = f32_to_bf16(val);
    }
  }
}

// ---------------------------------------------------------------------------
// host side
// ---------------------------------------------------------------------------
extern "C" void kernel_launch(void* const* d_in, const int* in_sizes, int n_in,
                              void* d_out, int out_size, void* d_ws,
                              size_t ws_size, hipStream_t stream) {
  (void)in_sizes; (void)n_in; (void)out_size; (void)ws_size;
  const float* x = (const float*)d_in[0];
  const float* w_q = (const float*)d_in[1];
  const float* w_k = (const float*)d_in[2];
  const float* w_v = (const float*)d_in[3];
  const float* w_o = (const float*)d_in[4];
  const float* q_gain = (const float*)d_in[5];
  const float* w_fc = (const float*)d_in[6];
  const float* w_proj = (const float*)d_in[7];
  const float* attn_scale = (const float*)d_in[8];
  const float* mlp_scale = (const float*)d_in[9];

  char* ws = (char*)d_ws;
  size_t off = 0;
  auto alloc = [&](size_t bytes) -> void* {
    off = (off + 255) & ~(size_t)255;
    void* p = ws + off;
    off += bytes;
    return p;
  };

  unsigned short* xn_b = (unsigned short*)alloc((size_t)MTOK * D_MODEL * 2);
  unsigned short* wq_b = (unsigned short*)alloc((size_t)D_MODEL * D_MODEL * 2);
  unsigned short* wk_b = (unsigned short*)alloc((size_t)KVD * D_MODEL * 2);
  unsigned short* wv_b = (unsigned short*)alloc((size_t)KVD * D_MODEL * 2);
  unsigned short* wo_b = (unsigned short*)alloc((size_t)D_MODEL * D_MODEL * 2);
  unsigned short* wfc_b = (unsigned short*)alloc((size_t)FDIM * D_MODEL * 2);
  unsigned short* wpr_b = (unsigned short*)alloc((size_t)D_MODEL * FDIM * 2);
  float* q_f = (float*)alloc((size_t)MTOK * D_MODEL * 4);  // reused as x2 later
  float* k_f = (float*)alloc((size_t)MTOK * KVD * 4);
  float* v_f = (float*)alloc((size_t)MTOK * KVD * 4);
  unsigned short* q_b = (unsigned short*)alloc((size_t)MTOK * D_MODEL * 2);
  unsigned short* k_b = (unsigned short*)alloc((size_t)MTOK * KVD * 2);
  unsigned short* v_b = (unsigned short*)alloc((size_t)MTOK * KVD * 2);
  unsigned short* y_b = (unsigned short*)alloc((size_t)MTOK * D_MODEL * 2);
  unsigned short* h_b = (unsigned short*)alloc((size_t)MTOK * FDIM * 2);

  auto cvt = [&](const float* src, unsigned short* dst, size_t n) {
    int n4 = (int)(n / 4);
    k_cvt_bf16<<<(n4 + 255) / 256, 256, 0, stream>>>(src, dst, n4);
  };
  cvt(w_q, wq_b, (size_t)D_MODEL * D_MODEL);
  cvt(w_k, wk_b, (size_t)KVD * D_MODEL);
  cvt(w_v, wv_b, (size_t)KVD * D_MODEL);
  cvt(w_o, wo_b, (size_t)D_MODEL * D_MODEL);
  cvt(w_fc, wfc_b, (size_t)FDIM * D_MODEL);
  cvt(w_proj, wpr_b, (size_t)D_MODEL * FDIM);

  // xn = rms_norm(x)
  k_rmsnorm_bf16<<<MTOK, 256, 0, stream>>>(x, xn_b, D_MODEL, 1e-6f);

  // q/k/v projections
  k_gemm_bf16<0><<<dim3(D_MODEL / 128, MTOK / 128), 256, 0, stream>>>(
      xn_b, wq_b, q_f, nullptr, nullptr, nullptr, MTOK, D_MODEL, D_MODEL);
  k_gemm_bf16<0><<<dim3(KVD / 128, MTOK / 128), 256, 0, stream>>>(
      xn_b, wk_b, k_f, nullptr, nullptr, nullptr, MTOK, KVD, D_MODEL);
  k_gemm_bf16<0><<<dim3(KVD / 128, MTOK / 128), 256, 0, stream>>>(
      xn_b, wv_b, v_f, nullptr, nullptr, nullptr, MTOK, KVD, D_MODEL);

  // qk-norm + rope (+gain/sqrt(d) folded into q); v relayout
  k_qkv_post<<<MTOK * NH / 8, 256, 0, stream>>>(q_f, q_b, q_gain, NH, 0);
  k_qkv_post<<<MTOK * NKV / 8, 256, 0, stream>>>(k_f, k_b, nullptr, NKV, 1);
  k_qkv_post<<<MTOK * NKV / 8, 256, 0, stream>>>(v_f, v_b, nullptr, NKV, 2);

  // causal flash attention
  k_attn<<<BATCH * NH * (SEQ / 128), 256, 0, stream>>>(q_b, k_b, v_b, y_b);

  // attn_out = y @ w_o^T ; x2 = x + attn_scale * attn_out   (q_f reused as x2)
  float* x2_f = q_f;
  k_gemm_bf16<2><<<dim3(D_MODEL / 128, MTOK / 128), 256, 0, stream>>>(
      y_b, wo_b, x2_f, nullptr, x, attn_scale, MTOK, D_MODEL, D_MODEL);

  // xn2 = rms_norm(x2)  (reuse xn_b)
  k_rmsnorm_bf16<<<MTOK, 256, 0, stream>>>(x2_f, xn_b, D_MODEL, 1e-6f);

  // h = relu(xn2 @ w_fc^T)^2 -> bf16
  k_gemm_bf16<1><<<dim3(FDIM / 128, MTOK / 128), 256, 0, stream>>>(
      xn_b, wfc_b, nullptr, h_b, nullptr, nullptr, MTOK, FDIM, D_MODEL);

  // out = x2 + mlp_scale * (h @ w_proj^T)
  k_gemm_bf16<2><<<dim3(D_MODEL / 128, MTOK / 128), 256, 0, stream>>>(
      h_b, wpr_b, (float*)d_out, nullptr, x2_f, mlp_scale, MTOK, D_MODEL, FDIM);
}
